// TargetEdgeInitializer_58334245814819
// MI455X (gfx1250) — matrix operand order repaired and checked by reading, not compiled, not run on hardware
//
#include <hip/hip_runtime.h>
#include <cmath>

// ---------------------------------------------------------------------------
// MI455X / gfx1250: wave32, WMMA 16x16x32 bf16 (f32 accumulate).
// LDS-staged GEMM: global_load_b128 -> v_cvt_pk_bf16_f32 -> ds_store_b128 ->
// ds_load_b128 fragments -> v_wmma_f32_16x16x32_bf16, with A-fragment reuse
// (2 WMMAs per wave per K-chunk). Branch-free K-loop via <TA,TB> templates.
// ---------------------------------------------------------------------------

typedef __attribute__((ext_vector_type(16))) __bf16 v16bf;
typedef __attribute__((ext_vector_type(8)))  __bf16 v8bf;
typedef __attribute__((ext_vector_type(4)))  __bf16 v4bf;
typedef __attribute__((ext_vector_type(8)))  float  v8f;

#define NNODES 1024
#define NEDGES 32768
#define NHEADS 8

// Monotone float<->uint encoding so atomicMin/Max on unsigned order floats.
__device__ __forceinline__ unsigned encf(float f) {
  unsigned u = __float_as_uint(f);
  return (u & 0x80000000u) ? ~u : (u | 0x80000000u);
}
__device__ __forceinline__ float decf(unsigned k) {
  return (k & 0x80000000u) ? __uint_as_float(k ^ 0x80000000u)
                           : __uint_as_float(~k);
}

__device__ __forceinline__ v8bf pack8(float4 f0, float4 f1) {
  v8bf pk;
  pk[0] = (__bf16)f0.x; pk[1] = (__bf16)f0.y;
  pk[2] = (__bf16)f0.z; pk[3] = (__bf16)f0.w;
  pk[4] = (__bf16)f1.x; pk[5] = (__bf16)f1.y;
  pk[6] = (__bf16)f1.z; pk[7] = (__bf16)f1.w;
  return pk;
}

// Build a WMMA fragment from two contiguous 8 x bf16 runs in LDS.
__device__ __forceinline__ v16bf frag16(const __bf16* p0, const __bf16* p1) {
  v8bf a0 = *(const v8bf*)p0;     // ds_load_b128
  v8bf a1 = *(const v8bf*)p1;
  return __builtin_shufflevector(a0, a1, 0, 1, 2, 3, 4, 5, 6, 7,
                                 8, 9, 10, 11, 12, 13, 14, 15);
}

// ---------------------------------------------------------------------------
// Wide GEMM: block tile 64(M) x 64(N); 8 waves; wave w -> (row w>>1, colpair
// w&1). Per K-chunk each wave: 1 A fragment (reused) + 2 B fragments ->
// 2 WMMAs. Requires M%64==0, N%64==0, K%32==0.
//   TA=0: A[m*lda+k]   TA=1: A[k*lda+m]
//   TB=0: B[k*ldb+n]   TB=1: B[n*ldb+k]
// ---------------------------------------------------------------------------
template <int TA, int TB>
__global__ void gemm64_bf16_wmma(const float* __restrict__ A,
                                 const float* __restrict__ B,
                                 const float* __restrict__ bias,
                                 float* __restrict__ C,
                                 int M, int N, int K,
                                 int lda, int ldb, int ldc,
                                 int doRelu, int accum)
{
  __shared__ __align__(16) __bf16 Alds[64 * 32];   // [m][k]
  __shared__ __align__(16) __bf16 Blds[64 * 32];   // [n][k]

  const int tid  = threadIdx.x;
  const int bN   = N >> 6;
  const int bm   = blockIdx.x / bN;
  const int bn   = blockIdx.x - bm * bN;
  const int m0   = bm * 64;
  const int n0   = bn * 64;
  const int wave = tid >> 5;
  const int lane = tid & 31;
  const int half = lane >> 4;
  const int l15  = lane & 15;
  const int wr   = wave >> 1;          // tile row 0..3
  const int wc   = wave & 1;           // 32-wide column pair 0..1

  v8f acc0 = {}, acc1 = {};

  for (int k0 = 0; k0 < K; k0 += 32) {
    __syncthreads();
    // ---- stage A panel: 64 m x 32 k ----
    if (TA == 0) {
      int r  = tid >> 2;
      int kq = (tid & 3) * 8;
      const float* src = A + (size_t)(m0 + r) * lda + (k0 + kq);
      if (k0 + 32 < K) __builtin_prefetch(src + 32, 0, 3);   // global_prefetch_b8
      float4 f0 = ((const float4*)src)[0];
      float4 f1 = ((const float4*)src)[1];
      *(v8bf*)&Alds[r * 32 + kq] = pack8(f0, f1);            // ds_store_b128
    } else {
      int kk = tid >> 3;
      int mq = (tid & 7) * 8;
      const float* src = A + (size_t)(k0 + kk) * lda + (m0 + mq);
      if (k0 + 32 < K) __builtin_prefetch(src + (size_t)32 * lda, 0, 3);
      float4 f0 = ((const float4*)src)[0];
      float4 f1 = ((const float4*)src)[1];
      float f[8] = {f0.x, f0.y, f0.z, f0.w, f1.x, f1.y, f1.z, f1.w};
#pragma unroll
      for (int i = 0; i < 8; ++i)
        Alds[(mq + i) * 32 + kk] = (__bf16)f[i];
    }
    // ---- stage B panel: 32 k x 64 n, column-major [n][k] ----
    if (TB == 0) {
      int kk = tid >> 3;
      int nq = (tid & 7) * 8;
      const float* src = B + (size_t)(k0 + kk) * ldb + (n0 + nq);
      if (k0 + 32 < K) __builtin_prefetch(src + (size_t)32 * ldb, 0, 3);
      float4 f0 = ((const float4*)src)[0];
      float4 f1 = ((const float4*)src)[1];
      float f[8] = {f0.x, f0.y, f0.z, f0.w, f1.x, f1.y, f1.z, f1.w};
#pragma unroll
      for (int i = 0; i < 8; ++i)
        Blds[(nq + i) * 32 + kk] = (__bf16)f[i];
    } else {
      int nn = tid >> 2;
      int kq = (tid & 3) * 8;
      const float* src = B + (size_t)(n0 + nn) * ldb + (k0 + kq);
      if (k0 + 32 < K) __builtin_prefetch(src + 32, 0, 3);
      float4 f0 = ((const float4*)src)[0];
      float4 f1 = ((const float4*)src)[1];
      *(v8bf*)&Blds[nn * 32 + kq] = pack8(f0, f1);           // ds_store_b128
    }
    __syncthreads();

    // A 16x32 layout: lanes 0-15 -> K 0..7 / 16..23 ; lanes 16-31 -> K 8..15 / 24..31
    const __bf16* arow = &Alds[(wr * 16 + l15) * 32 + half * 8];
    v16bf af = frag16(arow, arow + 16);
    // B 32x16 layout: lanes 0-15 -> K 0..7 / 16..23 ; lanes 16-31 -> K 8..15 / 24..31
    const __bf16* bc0 = &Blds[(wc * 32 + l15) * 32 + half * 8];
    const __bf16* bc1 = &Blds[(wc * 32 + 16 + l15) * 32 + half * 8];
    v16bf bf0 = frag16(bc0, bc0 + 16);
    v16bf bf1 = frag16(bc1, bc1 + 16);

    acc0 = __builtin_amdgcn_wmma_f32_16x16x32_bf16(
        false, af, false, bf0, (short)0, acc0, false, false);
    acc1 = __builtin_amdgcn_wmma_f32_16x16x32_bf16(
        false, af, false, bf1, (short)0, acc1, false, false);
  }

  // ---- epilogue: D layout lane=N, VGPR r -> M = r + 8*half ----
  int nA = n0 + wc * 32 + l15;
  int nB = nA + 16;
#pragma unroll
  for (int r = 0; r < 8; ++r) {
    int m = m0 + wr * 16 + r + half * 8;
    float v0 = acc0[r], v1 = acc1[r];
    if (bias) { v0 += bias[nA]; v1 += bias[nB]; }
    size_t i0 = (size_t)m * ldc + nA;
    size_t i1 = (size_t)m * ldc + nB;
    if (accum) { v0 += C[i0]; v1 += C[i1]; }
    if (doRelu) { v0 = fmaxf(v0, 0.0f); v1 = fmaxf(v1, 0.0f); }
    C[i0] = v0;
    C[i1] = v1;
  }
}

// ---------------------------------------------------------------------------
// Narrow GEMM for N%64!=0 (N%32==0): block tile 64 x 32, wave w -> tile
// (w>>1, w&1), one WMMA per wave per K-chunk.
// ---------------------------------------------------------------------------
template <int TA, int TB>
__global__ void gemm32_bf16_wmma(const float* __restrict__ A,
                                 const float* __restrict__ B,
                                 const float* __restrict__ bias,
                                 float* __restrict__ C,
                                 int M, int N, int K,
                                 int lda, int ldb, int ldc,
                                 int doRelu, int accum)
{
  __shared__ __align__(16) __bf16 Alds[64 * 32];   // [m][k]
  __shared__ __align__(16) __bf16 Blds[32 * 32];   // [n][k]

  const int tid  = threadIdx.x;
  const int bN   = N >> 5;
  const int bm   = blockIdx.x / bN;
  const int bn   = blockIdx.x - bm * bN;
  const int m0   = bm * 64;
  const int n0   = bn * 32;
  const int wave = tid >> 5;
  const int lane = tid & 31;
  const int half = lane >> 4;
  const int l15  = lane & 15;
  const int wr   = wave >> 1;
  const int wc   = wave & 1;

  v8f acc = {};

  for (int k0 = 0; k0 < K; k0 += 32) {
    __syncthreads();
    if (TA == 0) {
      int r  = tid >> 2;
      int kq = (tid & 3) * 8;
      const float* src = A + (size_t)(m0 + r) * lda + (k0 + kq);
      if (k0 + 32 < K) __builtin_prefetch(src + 32, 0, 3);
      float4 f0 = ((const float4*)src)[0];
      float4 f1 = ((const float4*)src)[1];
      *(v8bf*)&Alds[r * 32 + kq] = pack8(f0, f1);
    } else {
      int kk = tid >> 3;
      int mq = (tid & 7) * 8;
      const float* src = A + (size_t)(k0 + kk) * lda + (m0 + mq);
      if (k0 + 32 < K) __builtin_prefetch(src + (size_t)32 * lda, 0, 3);
      float4 f0 = ((const float4*)src)[0];
      float4 f1 = ((const float4*)src)[1];
      float f[8] = {f0.x, f0.y, f0.z, f0.w, f1.x, f1.y, f1.z, f1.w};
#pragma unroll
      for (int i = 0; i < 8; ++i)
        Alds[(mq + i) * 32 + kk] = (__bf16)f[i];
    }
    if (TB == 0) {
      int kk = tid >> 3;
      int nq = (tid & 7) * 4;
      const float* src = B + (size_t)(k0 + kk) * ldb + (n0 + nq);
      if (k0 + 32 < K) __builtin_prefetch(src + (size_t)32 * ldb, 0, 3);
      float4 f = *(const float4*)src;
      Blds[(nq + 0) * 32 + kk] = (__bf16)f.x;
      Blds[(nq + 1) * 32 + kk] = (__bf16)f.y;
      Blds[(nq + 2) * 32 + kk] = (__bf16)f.z;
      Blds[(nq + 3) * 32 + kk] = (__bf16)f.w;
    } else {
      int nn = tid >> 3;
      int kq = (tid & 7) * 4;
      const float* src = B + (size_t)(n0 + nn) * ldb + (k0 + kq);
      if (k0 + 32 < K) __builtin_prefetch(src + 32, 0, 3);
      float4 f = *(const float4*)src;
      v4bf pk;
      pk[0] = (__bf16)f.x; pk[1] = (__bf16)f.y;
      pk[2] = (__bf16)f.z; pk[3] = (__bf16)f.w;
      *(v4bf*)&Blds[nn * 32 + kq] = pk;
    }
    __syncthreads();

    const __bf16* arow = &Alds[(wr * 16 + l15) * 32 + half * 8];
    v16bf af = frag16(arow, arow + 16);
    const __bf16* bcol = &Blds[(wc * 16 + l15) * 32 + half * 8];
    v16bf bf = frag16(bcol, bcol + 16);

    acc = __builtin_amdgcn_wmma_f32_16x16x32_bf16(
        false, af, false, bf, (short)0, acc, false, false);
  }

  int n = n0 + wc * 16 + l15;
#pragma unroll
  for (int r = 0; r < 8; ++r) {
    int m = m0 + wr * 16 + r + half * 8;
    float v = acc[r];
    if (bias)   v += bias[n];
    size_t idx = (size_t)m * ldc + n;
    if (accum)  v += C[idx];
    if (doRelu) v = fmaxf(v, 0.0f);
    C[idx] = v;
  }
}

// ---------------------------------------------------------------------------
// Small utility kernels
// ---------------------------------------------------------------------------
__global__ void fill_f32(float* p, float v, int n) {
  for (int i = blockIdx.x * blockDim.x + threadIdx.x; i < n;
       i += gridDim.x * blockDim.x) p[i] = v;
}
__global__ void fill_u32(unsigned* p, unsigned v, int n) {
  for (int i = blockIdx.x * blockDim.x + threadIdx.x; i < n;
       i += gridDim.x * blockDim.x) p[i] = v;
}
__global__ void addvec(float* y, const float* a, int n) {
  for (int i = blockIdx.x * blockDim.x + threadIdx.x; i < n;
       i += gridDim.x * blockDim.x) y[i] += a[i];
}
__global__ void init_mm(unsigned* mm) {
  if (threadIdx.x == 0) {
    mm[0] = encf( 3.402823466e38f);   // min tracker (atomicMin)
    mm[1] = encf(-3.402823466e38f);   // max tracker (atomicMax)
  }
}
__global__ void minmax_reduce(const float* __restrict__ x, unsigned* mm, int n) {
  __shared__ float smin[256], smax[256];
  int tid = threadIdx.x;
  float lmin = 3.4e38f, lmax = -3.4e38f;
  for (int i = blockIdx.x * blockDim.x + tid; i < n;
       i += gridDim.x * blockDim.x) {
    float v = x[i];
    lmin = fminf(lmin, v);
    lmax = fmaxf(lmax, v);
  }
  smin[tid] = lmin; smax[tid] = lmax; __syncthreads();
  for (int o = 128; o; o >>= 1) {
    if (tid < o) {
      smin[tid] = fminf(smin[tid], smin[tid + o]);
      smax[tid] = fmaxf(smax[tid], smax[tid + o]);
    }
    __syncthreads();
  }
  if (tid == 0) {
    atomicMin(&mm[0], encf(smin[0]));
    atomicMax(&mm[1], encf(smax[0]));
  }
}
__global__ void minmax_norm(float* x, const unsigned* mm, int n) {
  float mn = decf(mm[0]), mx = decf(mm[1]);
  float inv = 1.0f / (mx - mn + 1e-8f);
  for (int i = blockIdx.x * blockDim.x + threadIdx.x; i < n;
       i += gridDim.x * blockDim.x) x[i] = (x[i] - mn) * inv;
}

// PyG GraphNorm + ReLU, in place; one block per channel.
__global__ void graph_norm_relu(float* __restrict__ X,
                                const float* __restrict__ w,
                                const float* __restrict__ b,
                                const float* __restrict__ ms,
                                int Nn, int Cc)
{
  int c = blockIdx.x, tid = threadIdx.x;
  __shared__ float red[256];
  __shared__ float bc0;
  float s = 0.f;
  for (int n = tid; n < Nn; n += blockDim.x) s += X[(size_t)n * Cc + c];
  red[tid] = s; __syncthreads();
  for (int o = 128; o; o >>= 1) { if (tid < o) red[tid] += red[tid + o]; __syncthreads(); }
  if (tid == 0) bc0 = red[0] / (float)Nn;
  __syncthreads();
  float m = ms[c] * bc0;
  __syncthreads();
  s = 0.f;
  for (int n = tid; n < Nn; n += blockDim.x) {
    float d = X[(size_t)n * Cc + c] - m; s += d * d;
  }
  red[tid] = s; __syncthreads();
  for (int o = 128; o; o >>= 1) { if (tid < o) red[tid] += red[tid + o]; __syncthreads(); }
  if (tid == 0) bc0 = red[0] / (float)Nn;
  __syncthreads();
  float scale = w[c] * rsqrtf(bc0 + 1e-5f);
  float bb = b[c];
  for (int n = tid; n < Nn; n += blockDim.x) {
    float d = X[(size_t)n * Cc + c] - m;
    X[(size_t)n * Cc + c] = fmaxf(scale * d + bb, 0.0f);
  }
}

// ---------------------------------------------------------------------------
// Edge-list attention (stage 1): segment softmax over incoming edges
// ---------------------------------------------------------------------------
__global__ void edge_alpha(const float* __restrict__ q, const float* __restrict__ k,
                           const float* __restrict__ ea, const float* __restrict__ We,
                           const int* __restrict__ ei, float* __restrict__ alp,
                           unsigned* __restrict__ amx, int E, int C, int hc, float invs)
{
  int t = blockIdx.x * blockDim.x + threadIdx.x;
  if (t >= E * NHEADS) return;
  int e = t / NHEADS, h = t - e * NHEADS;
  int src = ei[e], dst = ei[E + e];
  float attr = ea[e];
  float s = 0.f;
  for (int c = 0; c < C; ++c)
    s += q[(size_t)dst * hc + h * C + c] *
         (k[(size_t)src * hc + h * C + c] + attr * We[h * C + c]);
  s *= invs;
  alp[t] = s;
  atomicMax(&amx[dst * NHEADS + h], encf(s));
}
__global__ void edge_expsum(const float* __restrict__ alp, const unsigned* __restrict__ amx,
                            const int* __restrict__ ei, float* __restrict__ exb,
                            float* __restrict__ den, int E)
{
  int t = blockIdx.x * blockDim.x + threadIdx.x;
  if (t >= E * NHEADS) return;
  int e = t / NHEADS, h = t - e * NHEADS;
  int dst = ei[E + e];
  float ex = __expf(alp[t] - decf(amx[dst * NHEADS + h]));
  exb[t] = ex;
  atomicAdd(&den[dst * NHEADS + h], ex);
}
__global__ void edge_scatter(const float* __restrict__ exb, const float* __restrict__ den,
                             const float* __restrict__ v, const float* __restrict__ ea,
                             const float* __restrict__ We, const int* __restrict__ ei,
                             float* __restrict__ out, int E, int C, int hc)
{
  int t = blockIdx.x * blockDim.x + threadIdx.x;
  if (t >= E * NHEADS) return;
  int e = t / NHEADS, h = t - e * NHEADS;
  int src = ei[e], dst = ei[E + e];
  float a = exb[t] / (den[dst * NHEADS + h] + 1e-16f);
  float attr = ea[e];
  for (int c = 0; c < C; ++c)
    atomicAdd(&out[(size_t)dst * hc + h * C + c],
              (v[(size_t)src * hc + h * C + c] + attr * We[h * C + c]) * a);
}

// ---------------------------------------------------------------------------
// Dense attention (stage 2) helpers
// ---------------------------------------------------------------------------
__global__ void qw_kernel(const float* __restrict__ q, const float* __restrict__ We,
                          float* __restrict__ qw, int hc, int C)
{
  int t = blockIdx.x * blockDim.x + threadIdx.x;
  if (t >= NNODES * NHEADS) return;
  int j = t / NHEADS, h = t - j * NHEADS;
  float s = 0.f;
  for (int c = 0; c < C; ++c) s += q[(size_t)j * hc + h * C + c] * We[h * C + c];
  qw[t] = s;
}

// a[j,i] = softmax_i( (base[j,i] + qw[j,h]*A[i,j]) / sqrt(C) ); one block per row j.
__global__ void softmax_dense(const float* __restrict__ base, const float* __restrict__ qwv,
                              const float* __restrict__ Amat, float* __restrict__ aH,
                              int Nn, int h, float invs)
{
  int j = blockIdx.x, tid = threadIdx.x;
  __shared__ float red[256];
  __shared__ float bc0;
  float qwj = qwv[j * NHEADS + h];
  float lm = -3.4e38f;
  for (int i = tid; i < Nn; i += blockDim.x) {
    float s = (base[(size_t)j * Nn + i] + qwj * Amat[(size_t)i * Nn + j]) * invs;
    aH[(size_t)j * Nn + i] = s;
    lm = fmaxf(lm, s);
  }
  red[tid] = lm; __syncthreads();
  for (int o = 128; o; o >>= 1) { if (tid < o) red[tid] = fmaxf(red[tid], red[tid + o]); __syncthreads(); }
  if (tid == 0) bc0 = red[0];
  __syncthreads();
  float mx = bc0;
  __syncthreads();
  float ls = 0.f;
  for (int i = tid; i < Nn; i += blockDim.x) {
    float e = __expf(aH[(size_t)j * Nn + i] - mx);
    aH[(size_t)j * Nn + i] = e;
    ls += e;
  }
  red[tid] = ls; __syncthreads();
  for (int o = 128; o; o >>= 1) { if (tid < o) red[tid] += red[tid + o]; __syncthreads(); }
  if (tid == 0) bc0 = red[0];
  __syncthreads();
  float inv = 1.0f / bc0;
  for (int i = tid; i < Nn; i += blockDim.x) aH[(size_t)j * Nn + i] *= inv;
}

// rowA[j] = sum_i a[j,i]*A[i,j]; out[j, h*C + c] += rowA[j]*We[h*C+c]
__global__ void rowA_addw(const float* __restrict__ aH, const float* __restrict__ Amat,
                          const float* __restrict__ We, float* __restrict__ out,
                          int Nn, int h, int C, int hc)
{
  int j = blockIdx.x, tid = threadIdx.x;
  __shared__ float red[256];
  __shared__ float bc0;
  float s = 0.f;
  for (int i = tid; i < Nn; i += blockDim.x)
    s += aH[(size_t)j * Nn + i] * Amat[(size_t)i * Nn + j];
  red[tid] = s; __syncthreads();
  for (int o = 128; o; o >>= 1) { if (tid < o) red[tid] += red[tid + o]; __syncthreads(); }
  if (tid == 0) bc0 = red[0];
  __syncthreads();
  if (tid < C) out[(size_t)j * hc + h * C + tid] += bc0 * We[h * C + tid];
}

// Upper-triangular (k=1) extraction with min-max normalization fused.
__global__ void triu_extract(const float* __restrict__ Fm, const unsigned* __restrict__ mm,
                             float* __restrict__ out, int Nn)
{
  float mn = decf(mm[0]), mx = decf(mm[1]);
  float inv = 1.0f / (mx - mn + 1e-8f);
  for (long t = (long)blockIdx.x * blockDim.x + threadIdx.x; t < (long)Nn * Nn;
       t += (long)gridDim.x * blockDim.x) {
    int i = (int)(t / Nn), j = (int)(t % Nn);
    if (j <= i) continue;
    long pos = (long)i * (Nn - 1) - (long)i * (i - 1) / 2 + (j - i - 1);
    out[pos] = (Fm[t] - mn) * inv;
  }
}

// ---------------------------------------------------------------------------
// Host orchestration
// ---------------------------------------------------------------------------
struct ConvP {
  const float *We, *Wk, *Wq, *Ws, *Wv, *bk, *bq, *bs, *bv, *gnb, *gnms, *gnw;
};

extern "C" void kernel_launch(void* const* d_in, const int* in_sizes, int n_in,
                              void* d_out, int out_size, void* d_ws, size_t ws_size,
                              hipStream_t stream)
{
  (void)in_sizes; (void)n_in; (void)out_size; (void)ws_size;
  const int N = NNODES, E = NEDGES;

  const float* x  = (const float*)d_in[0];
  const int*   ei = (const int*)d_in[1];
  const float* ea = (const float*)d_in[2];
  // params flattened: sorted keys per level -> r1, r2, s1[0], s1[1], s2[0], s2[1]
  const float *r1W1 = (const float*)d_in[3], *r1W2 = (const float*)d_in[4];
  const float *r1b1 = (const float*)d_in[5], *r1b2 = (const float*)d_in[6];
  const float *r2W1 = (const float*)d_in[7], *r2W2 = (const float*)d_in[8];
  const float *r2b1 = (const float*)d_in[9], *r2b2 = (const float*)d_in[10];
  auto conv_at = [&](int i) {
    ConvP p;
    p.We  = (const float*)d_in[i + 0];  p.Wk   = (const float*)d_in[i + 1];
    p.Wq  = (const float*)d_in[i + 2];  p.Ws   = (const float*)d_in[i + 3];
    p.Wv  = (const float*)d_in[i + 4];  p.bk   = (const float*)d_in[i + 5];
    p.bq  = (const float*)d_in[i + 6];  p.bs   = (const float*)d_in[i + 7];
    p.bv  = (const float*)d_in[i + 8];  p.gnb  = (const float*)d_in[i + 9];
    p.gnms = (const float*)d_in[i + 10]; p.gnw = (const float*)d_in[i + 11];
    return p;
  };
  ConvP c10 = conv_at(11), c11 = conv_at(23), c20 = conv_at(35), c21 = conv_at(47);

  // ---- workspace carve-out (f32) ----
  float* W = (float*)d_ws;
  size_t o = 0;
  float* t1  = W + o; o += (size_t)N * 256;   // MLP hidden
  float* idb = W + o; o += (size_t)N * N;     // id1 / id2
  float* qb  = W + o; o += (size_t)N * N;
  float* kb  = W + o; o += (size_t)N * N;
  float* vb  = W + o; o += (size_t)N * N;
  float* ob  = W + o; o += (size_t)N * N;     // layer output A
  float* hb  = W + o; o += (size_t)N * N;     // layer output B
  float* Am  = W + o; o += (size_t)N * N;     // normalized Gram (persists stage 2)
  float* bs_ = W + o; o += (size_t)N * N;     // per-head base / final Fm
  float* aH  = W + o; o += (size_t)N * N;     // per-head attention
  float* alp = W + o; o += (size_t)E * NHEADS;
  float* exb = W + o; o += (size_t)E * NHEADS;
  float* qw  = W + o; o += (size_t)N * NHEADS;
  float* den = W + o; o += (size_t)N * NHEADS;
  unsigned* amx = (unsigned*)(W + o); o += (size_t)N * NHEADS;
  unsigned* mm  = (unsigned*)(W + o); o += 16;

  // tA/tB dispatch to branch-free template instantiations; 64-wide block tile
  // with A-fragment reuse when N%64==0, else 32-wide.
  auto gemm = [&](const float* A, const float* B, const float* bias, float* Cp,
                  int M, int Nc, int K, int lda, int ldb, int ldc,
                  int tA, int tB, int rl, int ac) {
    if ((Nc & 63) == 0) {
      int blocks = (M / 64) * (Nc / 64);
      if (tA == 0 && tB == 0)
        gemm64_bf16_wmma<0, 0><<<blocks, 256, 0, stream>>>(A, B, bias, Cp, M, Nc, K,
                                                           lda, ldb, ldc, rl, ac);
      else if (tA == 1)
        gemm64_bf16_wmma<1, 0><<<blocks, 256, 0, stream>>>(A, B, bias, Cp, M, Nc, K,
                                                           lda, ldb, ldc, rl, ac);
      else
        gemm64_bf16_wmma<0, 1><<<blocks, 256, 0, stream>>>(A, B, bias, Cp, M, Nc, K,
                                                           lda, ldb, ldc, rl, ac);
    } else {
      int blocks = (M / 64) * (Nc / 32);
      if (tA == 0 && tB == 0)
        gemm32_bf16_wmma<0, 0><<<blocks, 256, 0, stream>>>(A, B, bias, Cp, M, Nc, K,
                                                           lda, ldb, ldc, rl, ac);
      else if (tA == 1)
        gemm32_bf16_wmma<1, 0><<<blocks, 256, 0, stream>>>(A, B, bias, Cp, M, Nc, K,
                                                           lda, ldb, ldc, rl, ac);
      else
        gemm32_bf16_wmma<0, 1><<<blocks, 256, 0, stream>>>(A, B, bias, Cp, M, Nc, K,
                                                           lda, ldb, ldc, rl, ac);
    }
  };

  auto edge_layer = [&](const float* xin, int fin, int hc, const ConvP& p, float* dst) {
    int C = hc / NHEADS;
    float invs = 1.0f / sqrtf((float)C);
    gemm(xin, p.Wq, p.bq, qb,  N, hc, fin, fin, hc, hc, 0, 0, 0, 0);
    gemm(xin, p.Wk, p.bk, kb,  N, hc, fin, fin, hc, hc, 0, 0, 0, 0);
    gemm(xin, p.Wv, p.bv, vb,  N, hc, fin, fin, hc, hc, 0, 0, 0, 0);
    gemm(xin, p.Ws, p.bs, dst, N, hc, fin, fin, hc, hc, 0, 0, 0, 0);  // skip term
    int nh = N * NHEADS;
    fill_u32<<<(nh + 255) / 256, 256, 0, stream>>>(amx, 0x007FFFFFu, nh); // enc(-inf)
    fill_f32<<<(nh + 255) / 256, 256, 0, stream>>>(den, 0.0f, nh);
    int te = E * NHEADS;
    edge_alpha <<<(te + 255) / 256, 256, 0, stream>>>(qb, kb, ea, p.We, ei, alp, amx, E, C, hc, invs);
    edge_expsum<<<(te + 255) / 256, 256, 0, stream>>>(alp, amx, ei, exb, den, E);
    edge_scatter<<<(te + 255) / 256, 256, 0, stream>>>(exb, den, vb, ea, p.We, ei, dst, E, C, hc);
    graph_norm_relu<<<hc, 256, 0, stream>>>(dst, p.gnw, p.gnb, p.gnms, N, hc);
  };

  auto dense_layer = [&](const float* xin, int fin, int hc, const ConvP& p, float* dst) {
    int C = hc / NHEADS;
    float invs = 1.0f / sqrtf((float)C);
    gemm(xin, p.Wq, p.bq, qb,  N, hc, fin, fin, hc, hc, 0, 0, 0, 0);
    gemm(xin, p.Wk, p.bk, kb,  N, hc, fin, fin, hc, hc, 0, 0, 0, 0);
    gemm(xin, p.Wv, p.bv, vb,  N, hc, fin, fin, hc, hc, 0, 0, 0, 0);
    gemm(xin, p.Ws, p.bs, dst, N, hc, fin, fin, hc, hc, 0, 0, 0, 0);  // skip term
    qw_kernel<<<(N * NHEADS + 255) / 256, 256, 0, stream>>>(qb, p.We, qw, hc, C);
    for (int h = 0; h < NHEADS; ++h) {
      // base[j,i] = q_h[j,:] . k_h[i,:]   (K = C, B accessed transposed)
      gemm(qb + h * C, kb + h * C, nullptr, bs_, N, N, C, hc, hc, N, 0, 1, 0, 0);
      softmax_dense<<<N, 256, 0, stream>>>(bs_, qw, Am, aH, N, h, invs);
      // out[:, h*C:(h+1)*C] += aH @ v_h
      gemm(aH, vb + h * C, nullptr, dst + h * C, N, C, N, N, hc, hc, 0, 0, 0, 1);
      rowA_addw<<<N, 256, 0, stream>>>(aH, Am, p.We, dst, N, h, C, hc);
    }
    graph_norm_relu<<<hc, 256, 0, stream>>>(dst, p.gnw, p.gnb, p.gnms, N, hc);
  };

  // ---- stage 0: id1 = MLP(x) ----
  gemm(x,  r1W1, r1b1, t1,  N, 256, N,   N,   256, 256, 0, 0, 1, 0);
  gemm(t1, r1W2, r1b2, idb, N, N,   256, 256, N,   N,   0, 0, 0, 0);

  // ---- stage 1: two edge-list TransformerConv layers ----
  edge_layer(x,  N,   256, c10, ob);
  edge_layer(ob, 256, N,   c11, hb);
  addvec<<<(N * N + 255) / 256, 256, 0, stream>>>(hb, idb, N * N);   // h = h2 + id1

  // ---- A = minmax(h^T h) ----
  gemm(hb, hb, nullptr, Am, N, N, N, N, N, N, 1, 0, 0, 0);
  init_mm<<<1, 32, 0, stream>>>(mm);
  minmax_reduce<<<256, 256, 0, stream>>>(Am, mm, N * N);
  minmax_norm<<<256, 256, 0, stream>>>(Am, mm, N * N);

  // ---- id2 = MLP(A) ----
  gemm(Am, r2W1, r2b1, t1,  N, 256, N,   N,   256, 256, 0, 0, 1, 0);
  gemm(t1, r2W2, r2b2, idb, N, N,   256, 256, N,   N,   0, 0, 0, 0);

  // ---- stage 2: two dense TransformerConv layers ----
  dense_layer(Am, N,   256, c20, ob);
  dense_layer(ob, 256, N,   c21, hb);
  addvec<<<(N * N + 255) / 256, 256, 0, stream>>>(hb, idb, N * N);   // mid = mid + id2

  // ---- Fm = minmax(mid^T mid); emit upper triangle (k=1) ----
  gemm(hb, hb, nullptr, bs_, N, N, N, N, N, N, 1, 0, 0, 0);
  init_mm<<<1, 32, 0, stream>>>(mm);
  minmax_reduce<<<256, 256, 0, stream>>>(bs_, mm, N * N);
  triu_extract<<<2048, 256, 0, stream>>>(bs_, mm, (float*)d_out, N);
}
